// Gemma3Attention_6313601925401
// MI455X (gfx1250) — matrix operand order repaired and checked by reading, not compiled, not run on hardware
//
#include <hip/hip_runtime.h>

typedef __attribute__((ext_vector_type(16))) __bf16 v16bf;
typedef __attribute__((ext_vector_type(8)))  __bf16 v8bf;
typedef __attribute__((ext_vector_type(4)))  __bf16 v4bf;
typedef __attribute__((ext_vector_type(8)))  float  v8f;

#define B_    2
#define S_    2048
#define HID_  3072
#define NH_   24
#define NKV_  8
#define D_    128
#define G_    3
#define SCALE_ 0.08838834764831845f  /* 128^-0.5 */
#define CAP_   50.0f

static __device__ __forceinline__ v8f wmma_bf16(v16bf a, v16bf b, v8f c) {
  return __builtin_amdgcn_wmma_f32_16x16x32_bf16(false, a, false, b, (short)0, c,
                                                 false, false);
}

// CDNA5 async copy: 16B global -> 16B LDS, tracked by ASYNCcnt (GVS mode:
// saddr base + per-lane u32 byte offset). LDS dest = low 32 bits of generic ptr.
static __device__ __forceinline__ void async_b128(void* lds, const void* gbase,
                                                  unsigned byteOff) {
  asm volatile("global_load_async_to_lds_b128 %0, %1, %2"
               :
               : "v"((unsigned)(uintptr_t)lds), "v"(byteOff),
                 "s"((unsigned long long)(uintptr_t)gbase)
               : "memory");
}
static __device__ __forceinline__ void wait_async0() {
  asm volatile("s_wait_asynccnt 0x0" ::: "memory");
}

// A-fragment: lane holds row M=(lane&15); K = {8*(lane>>4)+0..7, 16+8*(lane>>4)+0..7}
static __device__ __forceinline__ v16bf ldsA(const __bf16* p) {
  v8bf lo = *(const v8bf*)p;
  v8bf hi = *(const v8bf*)(p + 16);
  v16bf a;
#pragma unroll
  for (int i = 0; i < 8; ++i) { a[i] = lo[i]; a[i + 8] = hi[i]; }
  return a;
}
// B-fragment: lane holds col N=(lane&15); K = {8*(lane>>4)+0..7, 16+8*(lane>>4)+0..7} (row stride ld)
static __device__ __forceinline__ v16bf ldsB(const __bf16* p, int ld) {
  v16bf b;
#pragma unroll
  for (int i = 0; i < 8; ++i) { b[i] = p[i * ld]; b[i + 8] = p[(16 + i) * ld]; }
  return b;
}

// ---------------------------------------------------------------- convert ----
__global__ void cvt_bf16(const float* __restrict__ s, __bf16* __restrict__ d,
                         long long n) {
  long long i = (long long)blockIdx.x * blockDim.x + threadIdx.x;
  const long long stride = (long long)gridDim.x * blockDim.x;
  for (; i < n; i += stride) d[i] = (__bf16)s[i];
}

// transpose + convert: src[N,K] f32 -> dst[K,N] bf16 (one-time weight prep so
// GEMM staging is pure contiguous async copies)
__global__ __launch_bounds__(256) void cvt_t_bf16(const float* __restrict__ src,
                                                  __bf16* __restrict__ dst,
                                                  int N, int K) {
  __shared__ float tile[32][33];
  const int nb = blockIdx.x * 32, kb = blockIdx.y * 32;
  const int tx = threadIdx.x & 31, ty = threadIdx.x >> 5;  // ty: 0..7
#pragma unroll
  for (int i = ty; i < 32; i += 8)
    tile[i][tx] = src[(size_t)(nb + i) * K + kb + tx];
  __syncthreads();
#pragma unroll
  for (int i = ty; i < 32; i += 8)
    dst[(size_t)(kb + i) * N + nb + tx] = (__bf16)tile[tx][i];
}

// ------------------------------------------------------------------- GEMM ----
// C[M,N] f32 = A[M,K] bf16 * Wt[K,N] bf16 ; double-buffered async staging.
__global__ __launch_bounds__(256) void gemm_bf16_kernel(
    const __bf16* __restrict__ A, const __bf16* __restrict__ Wt,
    float* __restrict__ C, int M, int N, int K) {
  __shared__ __bf16 As[2][128][48];    // row-major M x 32  (96B rows)
  __shared__ __bf16 Bs[2][32][144];    // k-major 32 x 128  (288B rows)
  const int t = threadIdx.x;
  const int lane = t & 31, wv = t >> 5;
  const int r = lane & 15, hf = lane >> 4;
  const int bm = blockIdx.y * 128, bn = blockIdx.x * 128;
  const int wm = (wv >> 1) * 32, wn = (wv & 1) * 64;

  const int arow = t >> 1, akb = (t & 1) * 16;  // A: 32B/thread
  const int brow = t >> 3, bnb = (t & 7) * 16;  // B: 32B/thread

  auto stage = [&](int buf, int k0) {
    const unsigned aoff = (unsigned)(((size_t)(bm + arow) * K + k0 + akb) * 2);
    async_b128(&As[buf][arow][akb], A, aoff);
    async_b128(&As[buf][arow][akb + 8], A, aoff + 16);
    const unsigned boff = (unsigned)(((size_t)(k0 + brow) * N + bn + bnb) * 2);
    async_b128(&Bs[buf][brow][bnb], Wt, boff);
    async_b128(&Bs[buf][brow][bnb + 8], Wt, boff + 16);
  };

  v8f acc[2][4] = {};
  stage(0, 0);
  const int nk = K / 32;
  for (int it = 0; it < nk; ++it) {
    const int cur = it & 1;
    wait_async0();
    __syncthreads();                       // tiles visible to all waves
    if (it + 1 < nk) stage(cur ^ 1, (it + 1) * 32);  // prefetch next tile

    v16bf af[2], bq[4];
#pragma unroll
    for (int mi = 0; mi < 2; ++mi)
      af[mi] = ldsA(&As[cur][wm + mi * 16 + r][hf * 8]);
    const int krow = r + hf * 16;
#pragma unroll
    for (int ni = 0; ni < 4; ++ni)
      bq[ni] = ldsB(&Bs[cur][hf * 8][wn + ni * 16 + r], 144);
#pragma unroll
    for (int mi = 0; mi < 2; ++mi)
#pragma unroll
      for (int ni = 0; ni < 4; ++ni)
        acc[mi][ni] = wmma_bf16(af[mi], bq[ni], acc[mi][ni]);
    __syncthreads();                       // done reading buf before overwrite
  }
#pragma unroll
  for (int mi = 0; mi < 2; ++mi)
#pragma unroll
    for (int ni = 0; ni < 4; ++ni)
#pragma unroll
      for (int v = 0; v < 8; ++v) {
        const int row = bm + wm + mi * 16 + v + 8 * hf;
        const int col = bn + wn + ni * 16 + r;
        C[(size_t)row * N + col] = acc[mi][ni][v];
      }
}

// --------------------------------------------------- RMSNorm + RoPE + pack ----
// src: [B*S, H*D] f32.  dMajor=0: dst[B*H, S, D]; dMajor=1: dst[B*H, D, S]
__global__ __launch_bounds__(256) void normrope_kernel(
    const float* __restrict__ src, __bf16* __restrict__ dst,
    const float* __restrict__ nw, const float* __restrict__ cosp,
    const float* __restrict__ sinp, int H, int doNormRope, int dMajor) {
  const int lane = threadIdx.x & 31;
  const int gw = blockIdx.x * 8 + (threadIdx.x >> 5);  // one wave per (bs,h)
  const int bs = gw / H, h = gw - bs * H;
  const int b = bs / S_, s = bs - b * S_;
  const int d0 = lane * 4;
  const float4 x4 = *(const float4*)(src + (size_t)bs * H * D_ + h * D_ + d0);
  float x[4] = {x4.x, x4.y, x4.z, x4.w};
  if (doNormRope) {
    float ss = x[0] * x[0] + x[1] * x[1] + x[2] * x[2] + x[3] * x[3];
#pragma unroll
    for (int off = 16; off >= 1; off >>= 1) ss += __shfl_xor(ss, off, 32);
    const float rinv = rsqrtf(ss * (1.0f / (float)D_) + 1e-6f);
    const float4 w4 = *(const float4*)(nw + d0);
    x[0] *= rinv * (1.0f + w4.x);
    x[1] *= rinv * (1.0f + w4.y);
    x[2] *= rinv * (1.0f + w4.z);
    x[3] *= rinv * (1.0f + w4.w);
    const float4 c4 = *(const float4*)(cosp + (size_t)bs * D_ + d0);
    const float4 s4 = *(const float4*)(sinp + (size_t)bs * D_ + d0);
    const float c[4] = {c4.x, c4.y, c4.z, c4.w};
    const float sn[4] = {s4.x, s4.y, s4.z, s4.w};
    const float sign = (lane < 16) ? -1.0f : 1.0f;  // d pairs with d^64 (xor 16)
#pragma unroll
    for (int i = 0; i < 4; ++i) {
      const float other = __shfl_xor(x[i], 16, 32);
      x[i] = x[i] * c[i] + sign * other * sn[i];
    }
  }
  if (dMajor) {  // K path: [b*H+h][d][s] so attention K-staging is contiguous
    const size_t base = (size_t)(b * H + h) * D_ * S_;
#pragma unroll
    for (int i = 0; i < 4; ++i)
      dst[base + (size_t)(d0 + i) * S_ + s] = (__bf16)x[i];
  } else {
    v4bf y;
#pragma unroll
    for (int i = 0; i < 4; ++i) y[i] = (__bf16)x[i];
    *(v4bf*)(dst + ((size_t)(b * H + h) * S_ + s) * D_ + d0) = y;
  }
}

// -------------------------------------------------------- flash attention ----
// Q: [B*NH, S, D], Kd: [B*NKV, D, S] (d-major), V: [B*NKV, S, D]
// O: [B*S, NH*D] bf16
__global__ __launch_bounds__(256) void attn_kernel(
    const __bf16* __restrict__ Q, const __bf16* __restrict__ Kd,
    const __bf16* __restrict__ Vc, __bf16* __restrict__ O) {
  __shared__ __bf16 Qs[128][144];      // q-rows x d               (36.9 KB)
  __shared__ __bf16 KTs[2][128][80];   // d-major [d][key] x2      (41.0 KB)
  __shared__ __bf16 Vs[2][64][144];    // [key][d] x2              (36.9 KB)
  __shared__ __bf16 Ps[128][80];       // probs [qrow][key]        (20.5 KB)
  const int t = threadIdx.x, lane = t & 31, wv = t >> 5;
  const int r = lane & 15, hf = lane >> 4;
  const int qt = blockIdx.x, h = blockIdx.y, b = blockIdx.z;
  const int kvh = h / G_;
  const size_t qoff = (size_t)(b * NH_ + h) * S_ * D_;
  const size_t koff = (size_t)(b * NKV_ + kvh) * D_ * S_;   // d-major
  const size_t voff = (size_t)(b * NKV_ + kvh) * S_ * D_;
  const int qbase = qt * 128;

  const int qrow = t >> 1, qch = (t & 1) * 64;   // Q: 128B/thread
  const int kdr = t >> 1, kkb = (t & 1) * 32;    // K: 64B/thread (row d, 64 keys)
  const int vkey = t >> 2, vdb = (t & 3) * 32;   // V: 64B/thread

  auto stageK = [&](int buf, int kbase) {
    const unsigned off = (unsigned)((koff + (size_t)kdr * S_ + kbase + kkb) * 2);
#pragma unroll
    for (int i = 0; i < 4; ++i)
      async_b128(&KTs[buf][kdr][kkb + i * 8], Kd, off + i * 16);
  };
  auto stageV = [&](int buf, int kbase) {
    const unsigned off = (unsigned)((voff + (size_t)(kbase + vkey) * D_ + vdb) * 2);
#pragma unroll
    for (int i = 0; i < 4; ++i)
      async_b128(&Vs[buf][vkey][vdb + i * 8], Vc, off + i * 16);
  };
  {  // Q tile: async once
    const unsigned off = (unsigned)((qoff + (size_t)(qbase + qrow) * D_ + qch) * 2);
#pragma unroll
    for (int i = 0; i < 8; ++i)
      async_b128(&Qs[qrow][qch + i * 8], Q, off + i * 16);
  }
  stageK(0, 0);
  stageV(0, 0);

  const int rbase = wv * 16;  // each wave owns 16 query rows exclusively
  float m_run[8], l_run[8];
  v8f oacc[8] = {};
#pragma unroll
  for (int v = 0; v < 8; ++v) { m_run[v] = -3.0e38f; l_run[v] = 0.0f; }

  const int ktEnd = qbase / 64 + 2;  // causal: skip tiles above the diagonal
  for (int kt = 0; kt < ktEnd; ++kt) {
    const int kbase = kt * 64;
    const int cur = kt & 1;
    wait_async0();
    __syncthreads();
    if (kt + 1 < ktEnd) { stageK(cur ^ 1, kbase + 64); stageV(cur ^ 1, kbase + 64); }

    // S16x64 = Q16x128 * K^T : 16 WMMA
    v8f sac[4] = {};
#pragma unroll
    for (int kc = 0; kc < 4; ++kc) {
      v16bf aq = ldsA(&Qs[rbase + r][kc * 32 + hf * 8]);
#pragma unroll
      for (int nt = 0; nt < 4; ++nt) {
        v16bf bk = ldsB(&KTs[cur][kc * 32 + hf * 8][nt * 16 + r], 80);
        sac[nt] = wmma_bf16(aq, bk, sac[nt]);
      }
    }

    // softcap + causal mask + online softmax (wave-private rows -> shuffles only)
    float mnew[8], lad[8];
#pragma unroll
    for (int v = 0; v < 8; ++v) {
      const int rowg = qbase + rbase + v + 8 * hf;
      float mx = -3.0e38f;
#pragma unroll
      for (int nt = 0; nt < 4; ++nt) {
        float sv = sac[nt][v] * SCALE_;
        sv = CAP_ * tanhf(sv * (1.0f / CAP_));
        const int colg = kbase + nt * 16 + r;
        if (colg > rowg) sv = -1.0e9f;
        sac[nt][v] = sv;
        mx = fmaxf(mx, sv);
      }
#pragma unroll
      for (int off = 1; off < 16; off <<= 1)
        mx = fmaxf(mx, __shfl_xor(mx, off, 32));
      mnew[v] = fmaxf(m_run[v], mx);
      float sum = 0.0f;
#pragma unroll
      for (int nt = 0; nt < 4; ++nt) {
        const float p = __expf(sac[nt][v] - mnew[v]);
        sac[nt][v] = p;
        sum += p;
      }
#pragma unroll
      for (int off = 1; off < 16; off <<= 1) sum += __shfl_xor(sum, off, 32);
      lad[v] = sum;
    }
#pragma unroll
    for (int v = 0; v < 8; ++v) {
      const float sc = __expf(m_run[v] - mnew[v]);
      l_run[v] = l_run[v] * sc + lad[v];
      m_run[v] = mnew[v];
#pragma unroll
      for (int dt = 0; dt < 8; ++dt) oacc[dt][v] = oacc[dt][v] * sc;
    }
    // C-frag -> A-frag relayout through LDS (wave-private rows)
#pragma unroll
    for (int nt = 0; nt < 4; ++nt)
#pragma unroll
      for (int v = 0; v < 8; ++v)
        Ps[rbase + v + 8 * hf][nt * 16 + r] = (__bf16)sac[nt][v];
    __syncthreads();

    // O16x128 += P16x64 * V64x128 : 16 WMMA
#pragma unroll
    for (int kp = 0; kp < 2; ++kp) {
      v16bf ap = ldsA(&Ps[rbase + r][kp * 32 + hf * 8]);
#pragma unroll
      for (int dt = 0; dt < 8; ++dt) {
        v16bf bv = ldsB(&Vs[cur][kp * 32 + hf * 8][dt * 16 + r], 144);
        oacc[dt] = wmma_bf16(ap, bv, oacc[dt]);
      }
    }
    __syncthreads();
  }

  float inv[8];
#pragma unroll
  for (int v = 0; v < 8; ++v) inv[v] = 1.0f / l_run[v];
#pragma unroll
  for (int dt = 0; dt < 8; ++dt)
#pragma unroll
    for (int v = 0; v < 8; ++v) {
      const int sIdx = qbase + rbase + v + 8 * hf;
      const int col = h * D_ + dt * 16 + r;
      O[(size_t)(b * S_ + sIdx) * (NH_ * D_) + col] =
          (__bf16)(oacc[dt][v] * inv[v]);
    }
}

// ----------------------------------------------------------------- launch ----
extern "C" void kernel_launch(void* const* d_in, const int* in_sizes, int n_in,
                              void* d_out, int out_size, void* d_ws,
                              size_t ws_size, hipStream_t stream) {
  const float* hidden = (const float*)d_in[0];
  const float* cosp   = (const float*)d_in[1];
  const float* sinp   = (const float*)d_in[2];
  /* d_in[3] attention_mask: pure causal -> computed analytically in-kernel */
  const float* Wq  = (const float*)d_in[4];
  const float* Wk  = (const float*)d_in[5];
  const float* Wv  = (const float*)d_in[6];
  const float* Wo  = (const float*)d_in[7];
  const float* qnw = (const float*)d_in[8];
  const float* knw = (const float*)d_in[9];
  (void)in_sizes; (void)n_in; (void)out_size; (void)ws_size;

  char* w = (char*)d_ws;
  size_t off = 0;
  auto take = [&](size_t bytes) -> char* {
    char* p = w + off;
    off = (off + bytes + 255) & ~(size_t)255;
    return p;
  };
  const size_t BS = (size_t)B_ * S_;  // 4096
  __bf16* Xbf  = (__bf16*)take(BS * HID_ * 2);
  __bf16* Wqt  = (__bf16*)take((size_t)NH_ * D_ * HID_ * 2);   // [K=HID][N=3072]
  __bf16* Wkt  = (__bf16*)take((size_t)NKV_ * D_ * HID_ * 2);  // [K=HID][N=1024]
  __bf16* Wvt  = (__bf16*)take((size_t)NKV_ * D_ * HID_ * 2);
  __bf16* Wot  = (__bf16*)take((size_t)HID_ * NH_ * D_ * 2);   // [K=3072][N=HID]
  float*  Qf   = (float*) take(BS * NH_ * D_ * 4);
  float*  Kf   = (float*) take(BS * NKV_ * D_ * 4);
  float*  Vf   = (float*) take(BS * NKV_ * D_ * 4);
  __bf16* Qbf  = (__bf16*)take(BS * NH_ * D_ * 2);
  __bf16* Kdm  = (__bf16*)take(BS * NKV_ * D_ * 2);  // d-major
  __bf16* Vbf  = (__bf16*)take(BS * NKV_ * D_ * 2);
  __bf16* Obf  = (__bf16*)take(BS * NH_ * D_ * 2);

  // staging: activations converted; weights converted + pre-transposed (k-major)
  cvt_bf16<<<2048, 256, 0, stream>>>(hidden, Xbf, (long long)BS * HID_);
  cvt_t_bf16<<<dim3(HID_ / 32, HID_ / 32), 256, 0, stream>>>(Wq, Wqt, HID_, HID_);
  cvt_t_bf16<<<dim3((NKV_ * D_) / 32, HID_ / 32), 256, 0, stream>>>(Wk, Wkt,
                                                                    NKV_ * D_, HID_);
  cvt_t_bf16<<<dim3((NKV_ * D_) / 32, HID_ / 32), 256, 0, stream>>>(Wv, Wvt,
                                                                    NKV_ * D_, HID_);
  cvt_t_bf16<<<dim3(HID_ / 32, (NH_ * D_) / 32), 256, 0, stream>>>(Wo, Wot,
                                                                   HID_, NH_ * D_);

  const dim3 blk(256);
  // QKV projections (async double-buffered WMMA GEMMs)
  gemm_bf16_kernel<<<dim3(HID_ / 128, BS / 128), blk, 0, stream>>>(
      Xbf, Wqt, Qf, (int)BS, NH_ * D_, HID_);
  gemm_bf16_kernel<<<dim3((NKV_ * D_) / 128, BS / 128), blk, 0, stream>>>(
      Xbf, Wkt, Kf, (int)BS, NKV_ * D_, HID_);
  gemm_bf16_kernel<<<dim3((NKV_ * D_) / 128, BS / 128), blk, 0, stream>>>(
      Xbf, Wvt, Vf, (int)BS, NKV_ * D_, HID_);

  // per-head RMSNorm + RoPE; K emitted d-major, V convert-only
  normrope_kernel<<<(unsigned)(BS * NH_ / 8), 256, 0, stream>>>(
      Qf, Qbf, qnw, cosp, sinp, NH_, 1, 0);
  normrope_kernel<<<(unsigned)(BS * NKV_ / 8), 256, 0, stream>>>(
      Kf, Kdm, knw, cosp, sinp, NKV_, 1, 1);
  normrope_kernel<<<(unsigned)(BS * NKV_ / 8), 256, 0, stream>>>(
      Vf, Vbf, nullptr, nullptr, nullptr, NKV_, 0, 0);

  // flash attention with tanh softcap + causal mask
  attn_kernel<<<dim3(S_ / 128, NH_, B_), blk, 0, stream>>>(Qbf, Kdm, Vbf, Obf);

  // output projection straight into d_out (f32)
  gemm_bf16_kernel<<<dim3(HID_ / 128, BS / 128), blk, 0, stream>>>(
      Obf, Wot, (float*)d_out, (int)BS, HID_, NH_ * D_);
}
